// AttnDecoder_45603962749812
// MI455X (gfx1250) — matrix operand order repaired and checked by reading, not compiled, not run on hardware
//
#include <hip/hip_runtime.h>
#include <utility>

// ---------------------------------------------------------------------------
// AttnDecoder for MI455X (gfx1250, wave32, WMMA).
// Attention path is fully linear => attn[b,t,:] = softmax_s(enc[b,s].u_enc),
// u_enc = W1[:,:H]^T (W2^T w3); dec terms cancel in softmax, ctx is per-batch.
// Heavy ops are f16-input/f32-acc WMMA GEMMs. The vocab GEMM (dominant HBM
// traffic, Wout ~65MB f16) uses 4 M-tiles per wave so each weight fragment is
// loaded once and reused by 4 WMMAs (4x lower weight traffic). The LSTM step
// kernel loads the hidden-state fragment once per k-step and reuses it across
// all 4 gate weight fragments.
// ---------------------------------------------------------------------------

typedef __attribute__((ext_vector_type(16))) _Float16 v16h;
typedef __attribute__((ext_vector_type(8)))  _Float16 v8h;
typedef __attribute__((ext_vector_type(8)))  float    v8f;

#define B_  8
#define T_  64
#define S_  128
#define H_  512
#define E_  512
#define V_  32000
#define G4H 2048   // 4*H
#define H3  1536   // 3*H
#define MT_ 4      // M-tiles per wave in the generic GEMM (64 rows)

// ---- WMMA fragment loaders -------------------------------------------------
// A 16x32 f16 fragment: lane holds row m=lane&15; khalf=lane>>4.
// elems 0..15 -> K = khalf*16 + j  (contiguous 32B, two 16B chunks)
__device__ __forceinline__ v16h load_a_frag(const _Float16* p /*row base + khalf*16*/) {
    v8h lo = *(const v8h*)p;
    v8h hi = *(const v8h*)(p + 8);
    v16h f;
#pragma unroll
    for (int j = 0; j < 8; ++j) { f[j] = lo[j]; f[j + 8] = hi[j]; }
    return f;
}
// B 32x16 f16 fragment: lane holds column n=lane&15; khalf=lane>>4.
// elems 0..15 -> K = khalf*16 + j  (one contiguous 32B chunk of the W row)
__device__ __forceinline__ v16h load_b_frag(const _Float16* p /*row base + khalf*16*/) {
    return *(const v16h*)p;
}

// ---- GEMM: C[M,N] = A[M,K](f16) * W[N,K](f16)^T + bias[N] -----------------
// grid = (N/16, M/64), block = 32 (one wave; 4 M-tiles x 1 N-tile).
// B fragment loaded once per k-step, reused by 4 WMMAs (weight-traffic / 4).
__global__ void gemm_wmma_f16(const _Float16* __restrict__ A,
                              const _Float16* __restrict__ W,
                              const float* __restrict__ bias,
                              float* __restrict__ C,
                              int M, int N, int K) {
    const int lane  = threadIdx.x;
    const int r     = lane & 15;
    const int khalf = lane >> 4;
    const int nt = blockIdx.x, mt = blockIdx.y;

    const _Float16* brow = W + (size_t)(nt * 16 + r) * K + khalf * 16;
    const _Float16* arow0 = A + (size_t)(mt * (16 * MT_) + r) * K + khalf * 16;

    v8f acc[MT_];
#pragma unroll
    for (int mi = 0; mi < MT_; ++mi) acc[mi] = (v8f){};

    for (int k0 = 0; k0 < K; k0 += 32) {
        if (k0 + 128 < K) {
            __builtin_prefetch(brow + k0 + 128, 0, 1);
            __builtin_prefetch(arow0 + k0 + 128, 0, 1);
        }
        v16h bv = load_b_frag(brow + k0);
#pragma unroll
        for (int mi = 0; mi < MT_; ++mi) {
            v16h av = load_a_frag(arow0 + (size_t)mi * 16 * K + k0);
            acc[mi] = __builtin_amdgcn_wmma_f32_16x16x32_f16(false, av, false, bv,
                                                             (short)0, acc[mi], false, false);
        }
    }
    const int col = nt * 16 + r;
    const float bb = bias ? bias[col] : 0.f;
#pragma unroll
    for (int mi = 0; mi < MT_; ++mi) {
        const int rbase = mt * (16 * MT_) + mi * 16 + khalf * 8;
#pragma unroll
        for (int j = 0; j < 8; ++j)
            C[(size_t)(rbase + j) * N + col] = acc[mi][j] + bb;
    }
}

// ---- LSTM recurrent step (one launch per timestep) -------------------------
// grid = H/16 = 32 blocks, block = 32 threads (one wave).
// k-outer loop: A (hidden state) fragment loaded once, reused by the 4 gate
// weight fragments -> 5 fragment loads per 4 WMMAs.
__global__ void lstm_step(const _Float16* __restrict__ hh_in,
                          _Float16* __restrict__ hh_out,
                          float* __restrict__ cst,
                          const _Float16* __restrict__ Whh,  // [4H, H] f16
                          const float* __restrict__ G,       // [B*T, 4H] f32
                          _Float16* __restrict__ ysh,        // [B*T, ldo] f16
                          int ldo, int t) {
    const int lane  = threadIdx.x;
    const int r     = lane & 15;
    const int khalf = lane >> 4;
    const int n0 = blockIdx.x * 16;

    const _Float16* arow = hh_in + (size_t)r * H_ + khalf * 16;
    const _Float16* brow = Whh + (size_t)(n0 + r) * H_ + khalf * 16;

    v8f acc[4];
#pragma unroll
    for (int q = 0; q < 4; ++q) acc[q] = (v8f){};

    for (int k0 = 0; k0 < H_; k0 += 32) {
        v16h av = load_a_frag(arow + k0);
#pragma unroll
        for (int q = 0; q < 4; ++q) {
            v16h bv = load_b_frag(brow + (size_t)q * H_ * H_ + k0);
            acc[q] = __builtin_amdgcn_wmma_f32_16x16x32_f16(false, av, false, bv,
                                                            (short)0, acc[q], false, false);
        }
    }

    const int col = n0 + r;          // D layout: lane&15 = column
    const int mbase = khalf * 8;     // rows mbase..mbase+7
#pragma unroll
    for (int j = 0; j < 8; ++j) {
        const int m = mbase + j;
        if (m < B_) {
            const int row = m * T_ + t;
            const float* g = G + (size_t)row * G4H;
            const float gi = acc[0][j] + g[col];
            const float gf = acc[1][j] + g[H_ + col];
            const float gg = acc[2][j] + g[2 * H_ + col];
            const float go = acc[3][j] + g[3 * H_ + col];
            const float si = 1.f / (1.f + expf(-gi));
            const float sf = 1.f / (1.f + expf(-gf));
            const float so = 1.f / (1.f + expf(-go));
            const float c  = sf * cst[m * H_ + col] + si * tanhf(gg);
            const float h  = so * tanhf(c);
            cst[m * H_ + col] = c;
            hh_out[m * H_ + col] = (_Float16)h;
            ysh[(size_t)row * ldo + col] = (_Float16)h;
        } else {
            hh_out[m * H_ + col] = (_Float16)0.f;  // keep padding rows zero
        }
    }
}

// ---- Small helper kernels --------------------------------------------------
__global__ void cvt_f16(const float* __restrict__ s, _Float16* __restrict__ d, int n) {
    int i = blockIdx.x * blockDim.x + threadIdx.x;
    if (i < n) d[i] = (_Float16)s[i];
}

__global__ void bias_sum(const float* a, const float* b, float* o, int n) {
    int i = blockIdx.x * blockDim.x + threadIdx.x;
    if (i < n) o[i] = a[i] + b[i];
}

__global__ void embed_gather(const int* __restrict__ ids, const float* __restrict__ emb,
                             _Float16* __restrict__ xh) {
    int i = blockIdx.x * blockDim.x + threadIdx.x;  // B*T*E
    if (i >= B_ * T_ * E_) return;
    int bt = i >> 9, e = i & (E_ - 1);
    xh[i] = (_Float16)emb[(size_t)ids[bt] * E_ + e];
}

__global__ void init_state(const float* __restrict__ h0, const float* __restrict__ c0,
                           _Float16* hhA, _Float16* hhB, float* cst) {
    int i = blockIdx.x * blockDim.x + threadIdx.x;  // 16*H
    if (i >= 16 * H_) return;
    int m = i >> 9, n = i & (H_ - 1);
    float hv = 0.f, cv = 0.f;
    if (m < B_) { hv = h0[m * H_ + n]; cv = c0[m * H_ + n]; }
    hhA[i] = (_Float16)hv;
    hhB[i] = (_Float16)hv;
    cst[i] = cv;
}

// v[g] = sum_f w3[f] * W2[f, g]   (W2 is [3H,3H] row-major)
__global__ void compute_v(const float* __restrict__ W2, const float* __restrict__ w3,
                          float* __restrict__ v) {
    int g = blockIdx.x * blockDim.x + threadIdx.x;
    if (g >= H3) return;
    float s = 0.f;
    for (int f = 0; f < H3; ++f) s += w3[f] * W2[(size_t)f * H3 + g];
    v[g] = s;
}

// u[h] = sum_g v[g] * W1[g, h]   (enc half of W1: columns 0..H-1, row stride 2H)
__global__ void compute_u(const float* __restrict__ W1, const float* __restrict__ v,
                          float* __restrict__ u) {
    int h = blockIdx.x * blockDim.x + threadIdx.x;
    if (h >= H_) return;
    float s = 0.f;
    for (int g = 0; g < H3; ++g) s += v[g] * W1[(size_t)g * (2 * H_) + h];
    u[h] = s;
}

// e[b,s] = enc[b,s,:] . u
__global__ void attn_scores(const float* __restrict__ enc, const float* __restrict__ u,
                            float* __restrict__ e) {
    int i = blockIdx.x * blockDim.x + threadIdx.x;  // B*S
    if (i >= B_ * S_) return;
    const float* er = enc + (size_t)i * H_;
    float s = 0.f;
    for (int h = 0; h < H_; ++h) s += er[h] * u[h];
    e[i] = s;
}

// per-b softmax over S, then ctx[b,:] = sum_s attn[s] * enc[b,s,:]
__global__ void softmax_ctx(const float* __restrict__ enc, const float* __restrict__ e,
                            float* __restrict__ ctx) {
    __shared__ float sm[S_];
    __shared__ float att[S_];
    const int b = blockIdx.x, s = threadIdx.x;
    const float ev = e[b * S_ + s];
    sm[s] = ev;
    __syncthreads();
    for (int off = S_ / 2; off > 0; off >>= 1) {
        if (s < off) sm[s] = fmaxf(sm[s], sm[s + off]);
        __syncthreads();
    }
    const float mx = sm[0];
    __syncthreads();
    const float ex = expf(ev - mx);
    sm[s] = ex;
    __syncthreads();
    for (int off = S_ / 2; off > 0; off >>= 1) {
        if (s < off) sm[s] += sm[s + off];
        __syncthreads();
    }
    att[s] = ex / sm[0];
    __syncthreads();
    for (int h = s; h < H_; h += S_) {
        float a = 0.f;
        for (int ss = 0; ss < S_; ++ss)
            a += att[ss] * enc[((size_t)b * S_ + ss) * H_ + h];
        ctx[b * H_ + h] = a;
    }
}

// feat[:,H:2H] = ctx[b,:] broadcast over t (f16)
__global__ void ctx_to_feat(const float* __restrict__ ctx, _Float16* __restrict__ feath) {
    int i = blockIdx.x * blockDim.x + threadIdx.x;  // B*T*H
    if (i >= B_ * T_ * H_) return;
    int bt = i >> 9, h = i & (H_ - 1);
    int b = bt / T_;
    feath[(size_t)bt * (2 * H_) + H_ + h] = (_Float16)ctx[b * H_ + h];
}

// ---------------------------------------------------------------------------
extern "C" void kernel_launch(void* const* d_in, const int* in_sizes, int n_in,
                              void* d_out, int out_size, void* d_ws, size_t ws_size,
                              hipStream_t stream) {
    const int*   ids  = (const int*)d_in[0];
    const float* enc  = (const float*)d_in[1];
    const float* h0   = (const float*)d_in[2];
    const float* c0   = (const float*)d_in[3];
    const float* emb  = (const float*)d_in[4];
    const float* Wih0 = (const float*)d_in[5];
    const float* Whh0 = (const float*)d_in[6];
    const float* bih0 = (const float*)d_in[7];
    const float* bhh0 = (const float*)d_in[8];
    const float* Wih1 = (const float*)d_in[9];
    const float* Whh1 = (const float*)d_in[10];
    const float* bih1 = (const float*)d_in[11];
    const float* bhh1 = (const float*)d_in[12];
    const float* W1   = (const float*)d_in[13];
    const float* W2   = (const float*)d_in[15];
    const float* w3   = (const float*)d_in[17];
    const float* Wout = (const float*)d_in[19];
    const float* bout = (const float*)d_in[20];
    float* out = (float*)d_out;

    // ---- workspace layout (256B aligned) ----
    char* base = (char*)d_ws;
    size_t off = 0;
    auto alloc = [&](size_t bytes) -> char* {
        char* r = base + off;
        off = (off + bytes + 255) & ~(size_t)255;
        return r;
    };
    _Float16* Wih0h = (_Float16*)alloc((size_t)G4H * E_ * 2);
    _Float16* Whh0h = (_Float16*)alloc((size_t)G4H * H_ * 2);
    _Float16* Wih1h = (_Float16*)alloc((size_t)G4H * H_ * 2);
    _Float16* Whh1h = (_Float16*)alloc((size_t)G4H * H_ * 2);
    _Float16* Wouth = (_Float16*)alloc((size_t)V_ * (2 * H_) * 2);
    _Float16* xh    = (_Float16*)alloc((size_t)B_ * T_ * E_ * 2);
    _Float16* ys0h  = (_Float16*)alloc((size_t)B_ * T_ * H_ * 2);
    _Float16* feath = (_Float16*)alloc((size_t)B_ * T_ * 2 * H_ * 2);
    float* G0    = (float*)alloc((size_t)B_ * T_ * G4H * 4);
    float* G1    = (float*)alloc((size_t)B_ * T_ * G4H * 4);
    float* bsum0 = (float*)alloc(G4H * 4);
    float* bsum1 = (float*)alloc(G4H * 4);
    float* vbuf  = (float*)alloc(H3 * 4);
    float* ubuf  = (float*)alloc(H_ * 4);
    float* ebuf  = (float*)alloc(B_ * S_ * 4);
    float* ctx   = (float*)alloc(B_ * H_ * 4);
    float* cst   = (float*)alloc(16 * H_ * 4);
    _Float16* hhA = (_Float16*)alloc(16 * H_ * 2);
    _Float16* hhB = (_Float16*)alloc(16 * H_ * 2);
    (void)ws_size; (void)in_sizes; (void)n_in; (void)out_size;

    const int TPB = 256;
    auto nb = [](int n, int t) { return (n + t - 1) / t; };

    // 1) weights -> f16
    cvt_f16<<<nb(G4H * E_, TPB), TPB, 0, stream>>>(Wih0, Wih0h, G4H * E_);
    cvt_f16<<<nb(G4H * H_, TPB), TPB, 0, stream>>>(Whh0, Whh0h, G4H * H_);
    cvt_f16<<<nb(G4H * H_, TPB), TPB, 0, stream>>>(Wih1, Wih1h, G4H * H_);
    cvt_f16<<<nb(G4H * H_, TPB), TPB, 0, stream>>>(Whh1, Whh1h, G4H * H_);
    cvt_f16<<<nb(V_ * 2 * H_, TPB), TPB, 0, stream>>>(Wout, Wouth, V_ * 2 * H_);
    bias_sum<<<nb(G4H, TPB), TPB, 0, stream>>>(bih0, bhh0, bsum0, G4H);
    bias_sum<<<nb(G4H, TPB), TPB, 0, stream>>>(bih1, bhh1, bsum1, G4H);

    // 2) embedding gather (f16)
    embed_gather<<<nb(B_ * T_ * E_, TPB), TPB, 0, stream>>>(ids, emb, xh);

    // 3) layer-0 input precompute: G0 = x @ Wih0^T + (bih0+bhh0)   [512 x 2048]
    gemm_wmma_f16<<<dim3(G4H / 16, (B_ * T_) / (16 * MT_)), 32, 0, stream>>>(
        xh, Wih0h, bsum0, G0, B_ * T_, G4H, E_);

    // 4) layer-0 recurrence (64 sequential steps, WMMA per step)
    init_state<<<nb(16 * H_, TPB), TPB, 0, stream>>>(h0, c0, hhA, hhB, cst);
    {
        _Float16 *ha = hhA, *hb = hhB;
        for (int t = 0; t < T_; ++t) {
            lstm_step<<<H_ / 16, 32, 0, stream>>>(ha, hb, cst, Whh0h, G0, ys0h, H_, t);
            std::swap(ha, hb);
        }
    }

    // 5) layer-1 input precompute: G1 = ys0 @ Wih1^T + (bih1+bhh1)
    gemm_wmma_f16<<<dim3(G4H / 16, (B_ * T_) / (16 * MT_)), 32, 0, stream>>>(
        ys0h, Wih1h, bsum1, G1, B_ * T_, G4H, H_);

    // 6) layer-1 recurrence; h written straight into feat[:, 0:H] (f16)
    init_state<<<nb(16 * H_, TPB), TPB, 0, stream>>>(h0 + B_ * H_, c0 + B_ * H_, hhA, hhB, cst);
    {
        _Float16 *ha = hhA, *hb = hhB;
        for (int t = 0; t < T_; ++t) {
            lstm_step<<<H_ / 16, 32, 0, stream>>>(ha, hb, cst, Whh1h, G1, feath, 2 * H_, t);
            std::swap(ha, hb);
        }
    }

    // 7) collapsed attention: u_enc, scores, softmax, per-batch context
    compute_v<<<nb(H3, TPB), TPB, 0, stream>>>(W2, w3, vbuf);
    compute_u<<<nb(H_, TPB), TPB, 0, stream>>>(W1, vbuf, ubuf);
    attn_scores<<<nb(B_ * S_, TPB), TPB, 0, stream>>>(enc, ubuf, ebuf);
    softmax_ctx<<<B_, S_, 0, stream>>>(enc, ebuf, ctx);
    ctx_to_feat<<<nb(B_ * T_ * H_, TPB), TPB, 0, stream>>>(ctx, feath);

    // 8) vocab projection: out[512, 32000] = feat @ Wout^T + bout (WMMA, f32 acc)
    gemm_wmma_f16<<<dim3(V_ / 16, (B_ * T_) / (16 * MT_)), 32, 0, stream>>>(
        feath, Wouth, bout, out, B_ * T_, V_, 2 * H_);
}
